// ProxyStudentTAttention_78245714199352
// MI455X (gfx1250) — matrix operand order repaired and checked by reading, not compiled, not run on hardware
//
#include <hip/hip_runtime.h>

typedef float    v8f  __attribute__((ext_vector_type(8)));
typedef _Float16 v16h __attribute__((ext_vector_type(16)));
typedef _Float16 h8   __attribute__((ext_vector_type(8)));
typedef _Float16 h4   __attribute__((ext_vector_type(4)));

namespace {
constexpr int B  = 2;
constexpr int T  = 2048;
constexpr int C  = 512;
constexpr int H  = 8;
constexpr int D  = 64;          // head dim
constexpr int BT = B * T;       // 4096
constexpr int BH = B * H;       // 16
}

__device__ __forceinline__ v16h cat16(h8 lo, h8 hi) {
  return __builtin_shufflevector(lo, hi, 0, 1, 2, 3, 4, 5, 6, 7,
                                         8, 9, 10, 11, 12, 13, 14, 15);
}

// ---------------------------------------------------------------------------
// Kernel 0: elementwise tanh (x activations and weight activations)
// ---------------------------------------------------------------------------
__global__ void tanh_kernel(const float* __restrict__ src, float* __restrict__ dst, int n) {
  int i = blockIdx.x * blockDim.x + threadIdx.x;
  if (i < n) dst[i] = tanhf(src[i]);
}

// ---------------------------------------------------------------------------
// Kernel 1: ProxyDeltaLinear forward == L1-cdist (forward of the proxy trick).
// out[n,j] = f((0.5 - (1/512) * sum_d |xact[n,d] - wact[j,d]|) * gain[j])
// f = tanh for q/k (which=0,1), identity for v (which=2).
// Written directly in (B*H, T, D) head-major layout.
// ---------------------------------------------------------------------------
__global__ __launch_bounds__(256) void qkv_l1_kernel(
    const float* __restrict__ xact,
    const float* __restrict__ wqa, const float* __restrict__ wka, const float* __restrict__ wva,
    const float* __restrict__ gq,  const float* __restrict__ gk,  const float* __restrict__ gv,
    float* __restrict__ qo, float* __restrict__ ko, float* __restrict__ vo) {
  const int nt    = blockIdx.x * 64;   // x-row tile base (0..4095)
  const int jt    = blockIdx.y * 64;   // output-feature tile base (0..511)
  const int which = blockIdx.z;        // 0=q 1=k 2=v

  const float* wact = (which == 0) ? wqa : (which == 1) ? wka : wva;
  const float* gain = (which == 0) ? gq  : (which == 1) ? gk  : gv;
  float*       dst  = (which == 0) ? qo  : (which == 1) ? ko  : vo;

  __shared__ __align__(16) float4 xs [64][9];   // 64 x 32 f32 (+pad: 36-dword stride)
  __shared__ __align__(16) float4 wsh[64][9];

  const int tid = threadIdx.x;
  const int tx  = tid & 15;       // cols tx + 16*jj
  const int ty  = tid >> 4;       // rows ty*4 + ii

  float acc[4][4];
#pragma unroll
  for (int i = 0; i < 4; ++i)
#pragma unroll
    for (int j = 0; j < 4; ++j) acc[i][j] = 0.0f;

  for (int k0 = 0; k0 < C; k0 += 32) {
    __syncthreads();
    for (int i = tid; i < 512; i += 256) {
      int r = i >> 3, c = i & 7;
      xs [r][c] = ((const float4*)(xact + (size_t)(nt + r) * C + k0))[c];
      wsh[r][c] = ((const float4*)(wact + (size_t)(jt + r) * C + k0))[c];
    }
    __syncthreads();

#pragma unroll
    for (int c = 0; c < 8; ++c) {
      float4 xv[4], wv4[4];
#pragma unroll
      for (int i = 0; i < 4; ++i) xv[i]  = xs[ty * 4 + i][c];   // broadcast across tx
#pragma unroll
      for (int j = 0; j < 4; ++j) wv4[j] = wsh[tx + j * 16][c]; // conflict-free stride
#pragma unroll
      for (int i = 0; i < 4; ++i)
#pragma unroll
        for (int j = 0; j < 4; ++j) {
          acc[i][j] += fabsf(xv[i].x - wv4[j].x) + fabsf(xv[i].y - wv4[j].y)
                     + fabsf(xv[i].z - wv4[j].z) + fabsf(xv[i].w - wv4[j].w);
        }
    }
  }

#pragma unroll
  for (int i = 0; i < 4; ++i) {
    const int n = nt + ty * 4 + i;
    const int b = n >> 11;          // n / T
    const int t = n & (T - 1);
#pragma unroll
    for (int j = 0; j < 4; ++j) {
      const int jc = jt + tx + j * 16;
      float l1m = acc[i][j] * (1.0f / (float)C);
      float val = (0.5f - l1m) * gain[jc];
      if (which != 2) val = tanhf(val);
      const int hh = jc >> 6;
      const int dd = jc & (D - 1);
      dst[((size_t)(b * H + hh) * T + t) * D + dd] = val;
    }
  }
}

// ---------------------------------------------------------------------------
// Kernel 2: fused causal proxy-Student-T attention.
// Per block: one (head, 64-query tile), 128 threads / 4 waves.
// Key tiles of 64 (skipping above-diagonal):
//   - L1 dist + kernel weight per (q,k); denominator via LDS ds_add_f32
//   - weights -> wt_s rows (A-fragment friendly), v -> vtT transposed
//     so fragments load as 2x ds_load_b128 per lane
//   - 2 K-steps x 4 col-groups of v_wmma_f32_16x16x32_f16 per wave per tile
// ---------------------------------------------------------------------------
__global__ __launch_bounds__(128) void attn_kernel(
    const float* __restrict__ qg_, const float* __restrict__ kg_, const float* __restrict__ vg_,
    const float* __restrict__ gamma, const float* __restrict__ rho,
    float* __restrict__ ao) {
  const int bh = blockIdx.x;            // 0..15
  const int qt = blockIdx.y;            // 0..31
  const int h  = bh & (H - 1);
  const int b  = bh >> 3;
  const int tid  = threadIdx.x;
  const int lane = tid & 31;
  const int wave = tid >> 5;

  __shared__ __align__(16) float    kt_s[64][64];   // key tile, f32 (16 KB)
  __shared__ __align__(16) _Float16 vtT [64][72];   // value tile TRANSPOSED [col][k]
  __shared__ __align__(16) _Float16 wt_s[64][72];   // weight tile [qrow][k]
  __shared__ float denom[64];

  const float gp = log1pf(__expf(gamma[h]));   // softplus
  const float rp = log1pf(__expf(rho[h]));

  const int q0      = qt * 64;
  const int qr      = tid & 63;         // this thread's query row (dist phase)
  const int halfSel = tid >> 6;         // which 32-key half
  const int qgidx   = q0 + qr;

  // q row in registers (16 float4 = 64 f32)
  float4 qv[16];
  {
    const float4* qp = (const float4*)(qg_ + ((size_t)bh * T + qgidx) * D);
#pragma unroll
    for (int i = 0; i < 16; ++i) qv[i] = qp[i];
  }

  if (tid < 64) denom[tid] = 0.0f;

  v8f acc[4];
#pragma unroll
  for (int g = 0; g < 4; ++g)
#pragma unroll
    for (int i = 0; i < 8; ++i) acc[g][i] = 0.0f;

  const int nkt = qt + 1;               // causal: 64-key tiles up to the diagonal

  for (int kt = 0; kt < nkt; ++kt) {
    const int kb = kt * 64;
    __syncthreads();  // previous iteration's WMMA reads complete before overwrite

    // stage k (f32) and v (f16, transposed); prefetch next tile into cache
    {
      const float* kp = kg_ + ((size_t)bh * T + kb) * D;
      for (int i = tid; i < 1024; i += 128) {          // 64x64 f32 as float4
        int r = i >> 4, c4 = i & 15;
        ((float4*)&kt_s[r][0])[c4] = ((const float4*)(kp + r * D))[c4];
        if (kt + 1 < nkt)
          __builtin_prefetch((const void*)(kp + 64 * D + r * D + c4 * 4), 0, 1);
      }
      const float* vp = vg_ + ((size_t)bh * T + kb) * D;
      for (int i = tid; i < 4096; i += 128) {
        int r = i >> 6, c = i & 63;
        vtT[c][r] = (_Float16)vp[r * D + c];
        if ((kt + 1 < nkt) && ((i & 63) == (tid & 63)))
          __builtin_prefetch((const void*)(vp + 64 * D + r * D + c), 0, 1);
      }
    }
    __syncthreads();

    // L1 distance + kernel weights for 32 keys of this thread's q row
    float part = 0.0f;
    const int kkbase = halfSel * 32;
#pragma unroll 1
    for (int kk = 0; kk < 32; ++kk) {
      const int kloc = kkbase + kk;
      const float4* kr = (const float4*)&kt_s[kloc][0];  // broadcast across lanes
      float sx = 0.f, sy = 0.f, sz = 0.f, sw = 0.f;
#pragma unroll
      for (int i = 0; i < 16; ++i) {
        float4 kv = kr[i];
        sx += fabsf(qv[i].x - kv.x);
        sy += fabsf(qv[i].y - kv.y);
        sz += fabsf(qv[i].z - kv.z);
        sw += fabsf(qv[i].w - kv.w);
      }
      const float s    = (sx + sy) + (sz + sw);
      const float ds   = s * 0.125f;                   // (s/64) * sqrt(64)
      const float ds2  = ds * ds;
      const float base = 1.0f / (1.0f + gp * ds2 + 1e-6f);
      float w = __powf(base, rp);
      if (kb + kloc > qgidx) w = 0.0f;                 // causal mask
      wt_s[qr][kloc] = (_Float16)w;
      part += w;
    }
    atomicAdd(&denom[qr], part);                        // ds_add_f32
    __syncthreads();

    // WMMA: wave owns query rows [wave*16, wave*16+16); 2 K-steps of 32
#pragma unroll
    for (int ks = 0; ks < 64; ks += 32) {
      const int r   = wave * 16 + (lane & 15);
      const int kb1 = (lane < 16) ? 0 : 8;
      h8 alo = *(const h8*)&wt_s[r][ks + kb1];
      h8 ahi = *(const h8*)&wt_s[r][ks + 16 + kb1];
      v16h afrag = cat16(alo, ahi);
#pragma unroll
      for (int g = 0; g < 4; ++g) {
        const int c0  = g * 16 + (lane & 15);
        const int kb2 = (lane < 16) ? 0 : 8;
        h8 blo = *(const h8*)&vtT[c0][ks + kb2];
        h8 bhi = *(const h8*)&vtT[c0][ks + 16 + kb2];
        v16h bfrag = cat16(blo, bhi);
        acc[g] = __builtin_amdgcn_wmma_f32_16x16x32_f16(
            false, afrag, false, bfrag, (short)0, acc[g], false, false);
      }
    }
  }
  __syncthreads();

  // normalize and scatter to (B, T, C) with head-interleave
  const int rbase = wave * 16 + ((lane < 16) ? 0 : 8);
  const int cl    = lane & 15;
#pragma unroll
  for (int g = 0; g < 4; ++g) {
#pragma unroll
    for (int i = 0; i < 8; ++i) {
      const int r   = rbase + i;
      const float d = denom[r] + 1e-6f;
      const int tq  = q0 + r;
      ao[((size_t)(b * T + tq)) * C + h * D + g * 16 + cl] = acc[g][i] / d;
    }
  }
}

// ---------------------------------------------------------------------------
// Kernel 3: output projection Y = AO @ Wo^T + bias via f16 WMMA.
// 64x64 output tile per 128-thread block (4 waves x 4 col-groups), K chunks 32.
// Fragment loads are 2x ds_load_b128 per lane (padded 40-half rows, 80 B stride).
// ---------------------------------------------------------------------------
__global__ __launch_bounds__(128) void proj_kernel(
    const float* __restrict__ ao, const float* __restrict__ wo,
    const float* __restrict__ bias, float* __restrict__ out) {
  const int nt = blockIdx.x * 64;       // row tile over 4096
  const int jt = blockIdx.y * 64;       // col tile over 512
  const int tid  = threadIdx.x;
  const int lane = tid & 31;
  const int wave = tid >> 5;

  __shared__ __align__(16) _Float16 at_s[64][40];   // A rows x K chunk
  __shared__ __align__(16) _Float16 wt_s[64][40];   // Wo rows (=out cols) x K chunk

  v8f acc[4];
#pragma unroll
  for (int g = 0; g < 4; ++g)
#pragma unroll
    for (int i = 0; i < 8; ++i) acc[g][i] = 0.0f;

  for (int k0 = 0; k0 < C; k0 += 32) {
    __syncthreads();
    for (int i = tid; i < 512; i += 128) {          // 64x32 elems as float4 -> h4
      int r = i >> 3, c4 = i & 7;
      float4 a4 = ((const float4*)(ao + (size_t)(nt + r) * C + k0))[c4];
      float4 w4 = ((const float4*)(wo + (size_t)(jt + r) * C + k0))[c4];
      *(h4*)&at_s[r][c4 * 4] = h4{(_Float16)a4.x, (_Float16)a4.y, (_Float16)a4.z, (_Float16)a4.w};
      *(h4*)&wt_s[r][c4 * 4] = h4{(_Float16)w4.x, (_Float16)w4.y, (_Float16)w4.z, (_Float16)w4.w};
    }
    __syncthreads();

    const int r   = wave * 16 + (lane & 15);
    const int kb1 = (lane < 16) ? 0 : 8;
    h8 alo = *(const h8*)&at_s[r][kb1];
    h8 ahi = *(const h8*)&at_s[r][16 + kb1];
    v16h afrag = cat16(alo, ahi);
#pragma unroll
    for (int g = 0; g < 4; ++g) {
      const int c0  = g * 16 + (lane & 15);      // local output column
      const int kb2 = (lane < 16) ? 0 : 8;
      h8 blo = *(const h8*)&wt_s[c0][kb2];       // B[k][n] = Wo[n][k], contiguous in k
      h8 bhi = *(const h8*)&wt_s[c0][16 + kb2];
      v16h bfrag = cat16(blo, bhi);
      acc[g] = __builtin_amdgcn_wmma_f32_16x16x32_f16(
          false, afrag, false, bfrag, (short)0, acc[g], false, false);
    }
  }

  const int rbase = wave * 16 + ((lane < 16) ? 0 : 8);
  const int cl    = lane & 15;
#pragma unroll
  for (int g = 0; g < 4; ++g) {
#pragma unroll
    for (int i = 0; i < 8; ++i) {
      const int r   = rbase + i;
      const int col = jt + g * 16 + cl;
      out[(size_t)(nt + r) * C + col] = acc[g][i] + bias[col];
    }
  }
}

// ---------------------------------------------------------------------------
extern "C" void kernel_launch(void* const* d_in, const int* in_sizes, int n_in,
                              void* d_out, int out_size, void* d_ws, size_t ws_size,
                              hipStream_t stream) {
  (void)in_sizes; (void)n_in; (void)out_size; (void)ws_size;

  const float* x     = (const float*)d_in[0];
  // d_in[1]: mask — structurally causal (tril) in the reference; exploited, not read.
  const float* wq    = (const float*)d_in[2];
  const float* gq    = (const float*)d_in[3];
  const float* wk    = (const float*)d_in[4];
  const float* gk    = (const float*)d_in[5];
  const float* wv    = (const float*)d_in[6];
  const float* gv    = (const float*)d_in[7];
  const float* wo    = (const float*)d_in[8];
  const float* wob   = (const float*)d_in[9];
  const float* gamma = (const float*)d_in[10];
  const float* rho   = (const float*)d_in[11];
  float* out = (float*)d_out;

  // workspace layout (floats)
  float* ws   = (float*)d_ws;
  float* xact = ws;                        // BT*C        = 2,097,152
  float* wqa  = xact + (size_t)BT * C;     // C*C         =   262,144
  float* wka  = wqa  + (size_t)C * C;
  float* wva  = wka  + (size_t)C * C;
  float* qb   = wva  + (size_t)C * C;      // BH*T*D      = 2,097,152
  float* kb   = qb   + (size_t)BH * T * D;
  float* vb   = kb   + (size_t)BH * T * D;
  float* ao   = vb   + (size_t)BH * T * D; // BT*C        = 2,097,152

  const int NX = BT * C;
  const int NW = C * C;
  tanh_kernel<<<(NX + 255) / 256, 256, 0, stream>>>(x,  xact, NX);
  tanh_kernel<<<(NW + 255) / 256, 256, 0, stream>>>(wq, wqa,  NW);
  tanh_kernel<<<(NW + 255) / 256, 256, 0, stream>>>(wk, wka,  NW);
  tanh_kernel<<<(NW + 255) / 256, 256, 0, stream>>>(wv, wva,  NW);

  dim3 g1(BT / 64, C / 64, 3);             // (64, 8, 3)
  qkv_l1_kernel<<<g1, 256, 0, stream>>>(xact, wqa, wka, wva, gq, gk, gv, qb, kb, vb);

  dim3 g2(BH, T / 64);                     // (16, 32)
  attn_kernel<<<g2, 128, 0, stream>>>(qb, kb, vb, gamma, rho, ao);

  dim3 g3(BT / 64, C / 64);                // (64, 8)
  proj_kernel<<<g3, 128, 0, stream>>>(ao, wo, wob, out);
}
